// DynamicShortConvolution_22016002359687
// MI455X (gfx1250) — matrix operand order repaired and checked by reading, not compiled, not run on hardware
//
#include <hip/hip_runtime.h>

// ---------------------------------------------------------------------------
// DynamicShortConvolution for MI455X (gfx1250, wave32, WMMA bf16)
//   GEMM1:  h  = silu(x @ w1^T)          [8192x2048] * [2048x2048]^T(row-major [N,K])
//   GEMM2:  k  = h @ w2_w^T + b          [8192x2048] * [8192x2048]^T(row-major [N,K])
//   fused epilogue: out[b,t,d] = silu( sum_w k[b,t,4d+w] * x[b,t+w-3,d] )
// Data movement: GLOBAL_LOAD_ASYNC_TO_LDS_B128 (ASYNCcnt) double-buffered tiles,
// no VGPR staging. Fast silu via v_exp_f32 + v_rcp_f32.
// ---------------------------------------------------------------------------

typedef __bf16 bf16;
typedef __attribute__((ext_vector_type(16))) __bf16 v16bf;
typedef __attribute__((ext_vector_type(8)))  __bf16 v8bf;
typedef __attribute__((ext_vector_type(8)))  float  v8f;

#define TILE_M 128
#define TILE_N 128
#define TILE_K 32
#define LDSTR  40      // ushort elements per LDS row: 32 K-values + 8 pad (80 B)

__device__ __forceinline__ float silu_f(float v) {
    // v * rcp(1 + exp(-v)) : v_exp_f32 + v_rcp_f32, no precise-div sequence
    return v * __builtin_amdgcn_rcpf(1.0f + __expf(-v));
}

__device__ __forceinline__ void wait_async0() {
#if __has_builtin(__builtin_amdgcn_s_wait_asynccnt)
    __builtin_amdgcn_s_wait_asynccnt(0);
#else
    asm volatile("s_wait_asynccnt 0x0" ::: "memory");
#endif
}

// Async global->LDS copy of one 16B chunk per lane (no VGPR data staging).
// LDS address operand = low 32 bits of the flat pointer (wave-relative LDS
// byte address per ISA 10.2; HW adds LDS_BASE).
__device__ __forceinline__ void async_b128(unsigned lds_addr, const void* gaddr) {
    asm volatile("global_load_async_to_lds_b128 %0, %1, off"
                 :: "v"(lds_addr), "v"(gaddr)
                 : "memory");
}

// Issue async loads for one 128x32 bf16 tile (row-major source, ld = K elems).
// 256 threads x 2 chunks x 16 B = 8 KB.
__device__ __forceinline__ void issue_tile_async(const bf16* __restrict__ g, int ld,
                                                 int r0, int k0, int tid,
                                                 unsigned short* sbase) {
    const int chunk = tid & 3;           // 4 x 16B = one 64B row of K
    const int row   = tid >> 2;          // 0..63 (+64 for second half)
    const char* g0 = (const char*)(g + (size_t)(r0 + row)      * ld + k0) + chunk * 16;
    const char* g1 = (const char*)(g + (size_t)(r0 + row + 64) * ld + k0) + chunk * 16;
    const unsigned l0 = (unsigned)(unsigned long long)(sbase + (size_t)row        * LDSTR)
                        + (unsigned)(chunk * 16);
    const unsigned l1 = (unsigned)(unsigned long long)(sbase + (size_t)(row + 64) * LDSTR)
                        + (unsigned)(chunk * 16);
    async_b128(l0, g0);
    async_b128(l1, g1);
}

// Fragment loads matching CDNA5 16-bit WMMA VGPR layouts (ISA 7.12.2):
//  A 16x32: lane<16 -> K 0..7 then 16..23 ; lane>=16 -> K 8..15 then 24..31
__device__ __forceinline__ v16bf load_frag_a(const unsigned short* s, int rowbase, int lane) {
    const int kh = lane >> 4;
    const int lr = lane & 15;
    const unsigned short* rp = s + (size_t)(rowbase + lr) * LDSTR;
    v8bf lo = *(const v8bf*)(rp + kh * 8);        // bytes [kh*16, +16)
    v8bf hi = *(const v8bf*)(rp + 16 + kh * 8);   // bytes [32+kh*16, +16)
    return __builtin_shufflevector(lo, hi, 0,1,2,3,4,5,6,7,8,9,10,11,12,13,14,15);
}
//  B 32x16: lane<16 -> K 0..7 then 16..23 ; lane>=16 -> K 8..15 then 24..31
__device__ __forceinline__ v16bf load_frag_b(const unsigned short* s, int rowbase, int lane) {
    const int kh = lane >> 4;
    const int lr = lane & 15;
    const unsigned short* rp = s + (size_t)(rowbase + lr) * LDSTR;
    v8bf lo = *(const v8bf*)(rp + kh * 8);
    v8bf hi = *(const v8bf*)(rp + 16 + kh * 8);
    return __builtin_shufflevector(lo, hi, 0,1,2,3,4,5,6,7,8,9,10,11,12,13,14,15);
}

// EPI = 0 : silu epilogue, store bf16 h
// EPI = 1 : bias + fused dynamic depthwise causal conv + silu, store f32 out
template <int EPI>
__global__ __launch_bounds__(256)
void gemm_bf16_wmma(const bf16* __restrict__ A,     // [M, K] bf16 row-major
                    const bf16* __restrict__ Bw,    // [N, K] bf16 row-major (used transposed)
                    int K, int N,
                    bf16* __restrict__ Hout,        // EPI==0: [M, N]
                    const float* __restrict__ bias, // EPI==1: [N]
                    const float* __restrict__ xf,   // EPI==1: [B,T,D] f32
                    float* __restrict__ out)        // EPI==1: [B,T,D] f32
{
    __shared__ unsigned short sA[2][TILE_M * LDSTR];
    __shared__ unsigned short sB[2][TILE_N * LDSTR];

    const int tid  = threadIdx.x;
    const int lane = tid & 31;
    const int wv   = tid >> 5;       // 8 waves
    const int wm   = wv & 3;         // 4 row groups  (32 rows each)
    const int wn   = wv >> 2;        // 2 col groups  (64 cols each)
    const int m0   = blockIdx.y * TILE_M;
    const int n0   = blockIdx.x * TILE_N;

    v8f acc[2][4];
#pragma unroll
    for (int i = 0; i < 2; ++i)
#pragma unroll
        for (int j = 0; j < 4; ++j)
            acc[i][j] = (v8f){0.f,0.f,0.f,0.f,0.f,0.f,0.f,0.f};

    // prologue: async-fill buffer 0
    issue_tile_async(A,  K, m0, 0, tid, sA[0]);
    issue_tile_async(Bw, K, n0, 0, tid, sB[0]);
    wait_async0();
    __syncthreads();

    const int nk = K / TILE_K;
    for (int kk = 0; kk < nk; ++kk) {
        const int cur = kk & 1;
        if (kk + 1 < nk) {
            // kick off next tile into the other buffer; async engine overlaps compute
            issue_tile_async(A,  K, m0, (kk + 1) * TILE_K, tid, sA[1 - cur]);
            issue_tile_async(Bw, K, n0, (kk + 1) * TILE_K, tid, sB[1 - cur]);
            if (kk + 2 < nk) {   // gfx1250 global_prefetch_b8 to warm L2 for k+2
                __builtin_prefetch((const char*)(A  + (size_t)(m0 + (tid >> 2)) * K + (kk + 2) * TILE_K), 0, 1);
                __builtin_prefetch((const char*)(Bw + (size_t)(n0 + (tid >> 2)) * K + (kk + 2) * TILE_K), 0, 1);
            }
        }

        v16bf af[2], bfr[4];
#pragma unroll
        for (int i = 0; i < 2; ++i)
            af[i] = load_frag_a(sA[cur], wm * 32 + i * 16, lane);
#pragma unroll
        for (int j = 0; j < 4; ++j)
            bfr[j] = load_frag_b(sB[cur], wn * 64 + j * 16, lane);

#pragma unroll
        for (int i = 0; i < 2; ++i)
#pragma unroll
            for (int j = 0; j < 4; ++j)
                acc[i][j] = __builtin_amdgcn_wmma_f32_16x16x32_bf16(
                    /*neg_a=*/false, af[i], /*neg_b=*/false, bfr[j],
                    /*c_mod=*/(short)0, acc[i][j],
                    /*reuse_a=*/false, /*reuse_b=*/false);

        if (kk + 1 < nk) {
            wait_async0();       // our async writes into buf[1-cur] landed
            __syncthreads();     // everyone done reading buf[cur] + all writes visible
        }
    }

    // ---------------- epilogue ----------------
    // C/D layout: lanes 0..15 -> N = base+lane, M = r ; lanes 16..31 -> same N set, M = r+8
    const int lr    = lane & 15;
    const int mhalf = (lane >> 4) << 3;

    if (EPI == 0) {
#pragma unroll
        for (int j = 0; j < 4; ++j) {
            const int n = n0 + wn * 64 + j * 16 + lr;
#pragma unroll
            for (int i = 0; i < 2; ++i) {
                const int mb = m0 + wm * 32 + i * 16 + mhalf;
#pragma unroll
                for (int r = 0; r < 8; ++r) {
                    float v = acc[i][j][r];
                    Hout[(size_t)(mb + r) * N + n] = (bf16)silu_f(v);
                }
            }
        }
    } else {
        // n = 4*d + w ; taps of one channel live in 4 adjacent lanes
#pragma unroll
        for (int j = 0; j < 4; ++j) {
            const int n  = n0 + wn * 64 + j * 16 + lr;
            const int w  = n & 3;
            const int d  = n >> 2;
            const float bv = bias[n];
#pragma unroll
            for (int i = 0; i < 2; ++i) {
                const int mb = m0 + wm * 32 + i * 16 + mhalf;
#pragma unroll
                for (int r = 0; r < 8; ++r) {
                    const int m = mb + r;            // global token row = b*4096 + t
                    const int t = m & 4095;
                    const int tt = t + w - 3;        // causal window, left-padded
                    float kv = acc[i][j][r] + bv;
                    float xv = (tt >= 0) ? xf[((size_t)(m - 3 + w)) * 2048 + d] : 0.0f;
                    float p  = kv * xv;
                    p += __shfl_xor(p, 1, 32);       // reduce the 4 taps across lanes
                    p += __shfl_xor(p, 2, 32);
                    if ((lane & 3) == 0)
                        out[(size_t)m * 2048 + d] = silu_f(p);
                }
            }
        }
    }
}

// f32 -> bf16 cast, 8 elements/thread (all sizes are multiples of 8)
__global__ __launch_bounds__(256)
void cvt_f32_to_bf16(const float* __restrict__ s, bf16* __restrict__ d, int n8) {
    int i = blockIdx.x * blockDim.x + threadIdx.x;
    if (i >= n8) return;
    const float4* p = (const float4*)s + (size_t)i * 2;
    float4 a = p[0], b = p[1];
    v8bf o;
    o[0] = (bf16)a.x; o[1] = (bf16)a.y; o[2] = (bf16)a.z; o[3] = (bf16)a.w;
    o[4] = (bf16)b.x; o[5] = (bf16)b.y; o[6] = (bf16)b.z; o[7] = (bf16)b.w;
    *((v8bf*)d + i) = o;
}

extern "C" void kernel_launch(void* const* d_in, const int* in_sizes, int n_in,
                              void* d_out, int out_size, void* d_ws, size_t ws_size,
                              hipStream_t stream) {
    (void)in_sizes; (void)n_in; (void)out_size; (void)ws_size;

    const float* x    = (const float*)d_in[0];  // [2,4096,2048]
    const float* w1   = (const float*)d_in[1];  // [2048,2048]
    const float* w2w  = (const float*)d_in[2];  // [8192,2048]
    const float* w2b  = (const float*)d_in[3];  // [8192]
    float*       out  = (float*)d_out;          // [2,4096,2048]

    const int M = 2 * 4096;   // 8192 tokens
    const int D = 2048;
    const int H = 2048;
    const int NW = D * 4;     // 8192

    char* ws = (char*)d_ws;
    bf16* xb   = (bf16*)(ws);                                   // 33554432 B
    bf16* w1b  = (bf16*)(ws + (size_t)33554432);                //  8388608 B
    bf16* w2wb = (bf16*)(ws + (size_t)41943040);                // 33554432 B
    bf16* hb   = (bf16*)(ws + (size_t)75497472);                // 33554432 B

    // 1) casts to bf16
    {
        int n8;
        n8 = (M * D) / 8;
        cvt_f32_to_bf16<<<(n8 + 255) / 256, 256, 0, stream>>>(x, xb, n8);
        n8 = (H * D) / 8;
        cvt_f32_to_bf16<<<(n8 + 255) / 256, 256, 0, stream>>>(w1, w1b, n8);
        n8 = (NW * H) / 8;
        cvt_f32_to_bf16<<<(n8 + 255) / 256, 256, 0, stream>>>(w2w, w2wb, n8);
    }

    // 2) GEMM1: h = silu(x @ w1^T)   -> bf16 h (stays resident in L2)
    {
        dim3 grid(H / TILE_N, M / TILE_M);   // (16, 64)
        gemm_bf16_wmma<0><<<grid, 256, 0, stream>>>(xb, w1b, D, H,
                                                    hb, nullptr, nullptr, nullptr);
    }

    // 3) GEMM2 + bias + fused dynamic causal conv + silu -> f32 out
    {
        dim3 grid(NW / TILE_N, M / TILE_M);  // (64, 64)
        gemm_bf16_wmma<1><<<grid, 256, 0, stream>>>(hb, w2wb, H, NW,
                                                    nullptr, w2b, x, out);
    }
}
